// EarthAttention3D_14010183319632
// MI455X (gfx1250) — matrix operand order repaired and checked by reading, not compiled, not run on hardware
//
#include <hip/hip_runtime.h>

#define NLAT 30
#define TOW 124
#define WD 144
#define DIM 192
#define HEADS 6
#define HD 32
#define NW 9                   // waves per block (9 * 16 = 144 rows)
#define NT (NW * 32)
#define QSCALE 0.17677669529663687f   // 1/sqrt(32)

typedef float v8f  __attribute__((ext_vector_type(8)));
typedef __bf16 v16bf __attribute__((ext_vector_type(16)));

union BFQ { v16bf v; uint4 q[2]; };

__device__ inline v8f vzero8() {
  v8f z = {0.f, 0.f, 0.f, 0.f, 0.f, 0.f, 0.f, 0.f};
  return z;
}

__device__ inline v8f wmma_bf16(v16bf a, v16bf b, v8f c) {
  // D = A(16x32 bf16) * B(32x16 bf16) + C(16x16 f32)
  return __builtin_amdgcn_wmma_f32_16x16x32_bf16(
      false, a, false, b, (short)0, c, false, false);
}

// ---- fragment loaders (CDNA5 ISA 7.12.2, wave32). All LDS strides are
// multiples of 8 halfs so every chunk below is 16B-aligned -> ds_load_b128.

// A (16x32 bf16) from row-major LDS [16][ld]
__device__ inline v16bf load_a_lds(const __bf16* p0, int ld, int lane) {
  const int r = lane & 15, kb = (lane >> 4) * 8;
  const __bf16* p = p0 + r * ld + kb;
  BFQ t;
  t.q[0] = *(const uint4*)(p);        // K kb..kb+7
  t.q[1] = *(const uint4*)(p + 16);   // K kb+16..kb+23
  return t.v;
}
// B (32x16 bf16) from transposed LDS storage [N][ld]: same K order as A
__device__ inline v16bf load_bT_lds(const __bf16* p0, int ld, int lane) {
  const int n = lane & 15, kb = (lane >> 4) * 8;
  const __bf16* p = p0 + n * ld + kb;
  BFQ t;
  t.q[0] = *(const uint4*)(p);        // K kb..kb+7
  t.q[1] = *(const uint4*)(p + 16);   // K kb+16..kb+23
  return t.v;
}
// A (16x32) straight from global f32, converting to bf16 (vectorized b128 loads)
__device__ inline v16bf load_a_gf32(const float* __restrict__ p0, int ld, int lane) {
  const int r = lane & 15, kb = (lane >> 4) * 8;
  const float* p = p0 + r * ld + kb;
  float f[16];
  *(float4*)(f + 0)  = *(const float4*)(p + 0);
  *(float4*)(f + 4)  = *(const float4*)(p + 4);
  *(float4*)(f + 8)  = *(const float4*)(p + 16);
  *(float4*)(f + 12) = *(const float4*)(p + 20);
  v16bf t;
#pragma unroll
  for (int e = 0; e < 16; ++e) t[e] = (__bf16)f[e];
  return t;
}

// ---- LDS pool layout (bytes). The w_qkv slice time-shares the P region; the
// w_out staging reuses the whole pool after the head loop. All offsets 16B-aligned.
#define OFF_QS   0                        // [NW][16][32]      9,216 B
#define OFF_KS   9216                     // [144][40]        11,520 B
#define OFF_VST  20736                    // [32][168] (V^T)  10,752 B
#define OFF_PS   31488                    // [NW][16][168]    48,384 B
#define OFF_WQL  31488                    // [96][104] (w^T)  19,968 B (union w/ PS)
#define OFF_WOL  0                        // [192][200] (w^T) 76,800 B (union w/ all)
#define POOL_BYTES 79872

__global__ __launch_bounds__(NT)
void earth_attn3d_kernel(const float* __restrict__ x,
                         const float* __restrict__ mask,
                         const float* __restrict__ w_qkv,
                         const float* __restrict__ b_qkv,
                         const float* __restrict__ w_out,
                         const float* __restrict__ b_out,
                         const float* __restrict__ ebias,
                         float* __restrict__ out) {
  __shared__ alignas(16) unsigned char pool[POOL_BYTES];
  __shared__ alignas(16) __bf16 os_s[WD * 200];     // 57,600 B, persists all heads

  __bf16* qs   = (__bf16*)(pool + OFF_QS);
  __bf16* ksm  = (__bf16*)(pool + OFF_KS);
  __bf16* vsT  = (__bf16*)(pool + OFF_VST);
  __bf16* ps   = (__bf16*)(pool + OFF_PS);
  __bf16* wqlT = (__bf16*)(pool + OFF_WQL);
  __bf16* wolT = (__bf16*)(pool + OFF_WOL);

  const int blk  = blockIdx.x;
  const int nl   = blk / TOW;
  const int tw   = blk % TOW;
  const int tid  = threadIdx.x;
  const int wave = tid >> 5;
  const int lane = tid & 31;
  const int half = lane >> 4;
  const int ln   = lane & 15;
  const int m0   = wave * 16;

  const float* xw   = x    + (size_t)(nl * TOW + tw) * WD * DIM;
  float*       ow   = out  + (size_t)(nl * TOW + tw) * WD * DIM;
  const float* mrow = mask + (size_t)(nl * TOW + tw) * WD * WD;

  // zero V^T pad columns (tokens 144..159) once; region never overwritten by staging
  for (int idx = tid; idx < 32 * 16; idx += NT)
    vsT[(idx >> 4) * 168 + 144 + (idx & 15)] = (__bf16)0.0f;

  // head-invariant x A-fragments for this wave's 16-row strip
  v16bf ax[6];
#pragma unroll
  for (int kc = 0; kc < 6; ++kc)
    ax[kc] = load_a_gf32(xw + m0 * DIM + kc * 32, DIM, lane);

  for (int h = 0; h < HEADS; ++h) {
    __syncthreads();   // previous head's PV reads of vsT/ps (and wqlT region) done

    // ---- stage w_qkv head-slice into LDS, transposed [c][k], bf16 ----------
    // c in [0,96): 0..31 -> Q cols, 32..63 -> K cols, 64..95 -> V cols
    for (int idx = tid; idx < 96 * 192; idx += NT) {
      const int c = idx % 96;            // fast index -> coalesced global reads
      const int k = idx / 96;
      const int matc = c >> 5, cc = c & 31;
      wqlT[c * 104 + k] =
          (__bf16)w_qkv[(size_t)k * (3 * DIM) + matc * DIM + h * HD + cc];
    }
    __syncthreads();

    // ---- QKV projection: 6 output tiles (Q0 Q1 K0 K1 V0 V1) ----------------
    v8f acc[6];
#pragma unroll
    for (int j = 0; j < 6; ++j) acc[j] = vzero8();
#pragma unroll
    for (int kc = 0; kc < 6; ++kc) {
#pragma unroll
      for (int j = 0; j < 6; ++j) {
        v16bf b = load_bT_lds(wqlT + (j * 16) * 104 + kc * 32, 104, lane);
        acc[j] = wmma_bf16(ax[kc], b, acc[j]);
      }
    }

    // deposit Q (scaled) / K / V^T into LDS with bias add
#pragma unroll
    for (int j = 0; j < 6; ++j) {
      const int mat = j >> 1, sub = j & 1;
      const float bb = b_qkv[mat * DIM + h * HD + sub * 16 + ln];
#pragma unroll
      for (int i = 0; i < 8; ++i) {
        const int r = i + half * 8;
        const float v = acc[j][i] + bb;
        if (mat == 0)      qs[wave * 512 + r * 32 + sub * 16 + ln] = (__bf16)(v * QSCALE);
        else if (mat == 1) ksm[(m0 + r) * 40 + sub * 16 + ln]      = (__bf16)v;
        else               vsT[(sub * 16 + ln) * 168 + (m0 + r)]   = (__bf16)v;
      }
    }
    __syncthreads();

    // ---- S = Q * K^T (K-dim = 32: one WMMA per 16x16 tile) -----------------
    v16bf aq = load_a_lds(qs + wave * 512, 32, lane);
    v8f s[9];
#pragma unroll
    for (int ct = 0; ct < 9; ++ct) {
      v16bf bk = load_bT_lds(ksm + (ct * 16) * 40, 40, lane);
      s[ct] = wmma_bf16(aq, bk, vzero8());
    }

    // bias + mask (L2-resident slices)
#pragma unroll
    for (int ct = 0; ct < 9; ++ct) {
#pragma unroll
      for (int i = 0; i < 8; ++i) {
        const int q   = m0 + i + half * 8;
        const int key = ct * 16 + ln;
        s[ct][i] += ebias[(((size_t)tw * HEADS + h) * WD + q) * WD + key]
                  + mrow[(size_t)q * WD + key];
      }
    }

    // softmax: each row lives entirely in one 16-lane half
#pragma unroll
    for (int i = 0; i < 8; ++i) {
      float rm = -3.4e38f;
#pragma unroll
      for (int ct = 0; ct < 9; ++ct) rm = fmaxf(rm, s[ct][i]);
#pragma unroll
      for (int msk = 1; msk < 16; msk <<= 1) rm = fmaxf(rm, __shfl_xor(rm, msk, 16));
      float rs = 0.f;
#pragma unroll
      for (int ct = 0; ct < 9; ++ct) { float e = __expf(s[ct][i] - rm); s[ct][i] = e; rs += e; }
#pragma unroll
      for (int msk = 1; msk < 16; msk <<= 1) rs += __shfl_xor(rs, msk, 16);
      const float inv = 1.f / rs;
#pragma unroll
      for (int ct = 0; ct < 9; ++ct) s[ct][i] *= inv;
    }

    // P -> LDS strip (bf16) + re-zero this strip's K-pad (wqlT staging clobbers it)
#pragma unroll
    for (int ct = 0; ct < 9; ++ct)
#pragma unroll
      for (int i = 0; i < 8; ++i)
        ps[wave * 2688 + (i + half * 8) * 168 + ct * 16 + ln] = (__bf16)s[ct][i];
#pragma unroll
    for (int i = 0; i < 8; ++i)
      ps[wave * 2688 + (i + half * 8) * 168 + 144 + ln] = (__bf16)0.0f;

    // ---- O = P * V (K padded 144 -> 160: 5 WMMA chunks) --------------------
    v8f o0 = vzero8(), o1 = vzero8();
#pragma unroll
    for (int kc = 0; kc < 5; ++kc) {
      v16bf ap = load_a_lds(ps + wave * 2688 + kc * 32, 168, lane);
      v16bf b0 = load_bT_lds(vsT + kc * 32, 168, lane);            // n0 = 0
      v16bf b1 = load_bT_lds(vsT + 16 * 168 + kc * 32, 168, lane); // n0 = 16
      o0 = wmma_bf16(ap, b0, o0);
      o1 = wmma_bf16(ap, b1, o1);
    }

    // stash head output into os[:, h*32 .. h*32+31]
#pragma unroll
    for (int i = 0; i < 8; ++i) {
      const int r = m0 + i + half * 8;
      os_s[r * 200 + h * HD + ln]      = (__bf16)o0[i];
      os_s[r * 200 + h * HD + 16 + ln] = (__bf16)o1[i];
    }
  }

  __syncthreads();

  // ---- stage w_out transposed [c][k] into the (now dead) pool --------------
  for (int idx = tid; idx < 192 * 192; idx += NT) {
    const int c = idx % 192;             // fast index -> coalesced global reads
    const int k = idx / 192;
    wolT[c * 200 + k] = (__bf16)w_out[(size_t)k * DIM + c];
  }
  __syncthreads();

  // ---- out = os @ w_out + b_out --------------------------------------------
  v16bf ao[6];
#pragma unroll
  for (int kc = 0; kc < 6; ++kc)
    ao[kc] = load_a_lds(os_s + m0 * 200 + kc * 32, 200, lane);

#pragma unroll
  for (int nt = 0; nt < 12; ++nt) {
    v8f acc = vzero8();
#pragma unroll
    for (int kc = 0; kc < 6; ++kc) {
      v16bf b = load_bT_lds(wolT + (nt * 16) * 200 + kc * 32, 200, lane);
      acc = wmma_bf16(ao[kc], b, acc);
    }
    const float bb = b_out[nt * 16 + ln];
#pragma unroll
    for (int i = 0; i < 8; ++i)
      ow[(size_t)(m0 + i + half * 8) * DIM + nt * 16 + ln] = acc[i] + bb;
  }
}

extern "C" void kernel_launch(void* const* d_in, const int* in_sizes, int n_in,
                              void* d_out, int out_size, void* d_ws, size_t ws_size,
                              hipStream_t stream) {
  (void)in_sizes; (void)n_in; (void)out_size; (void)d_ws; (void)ws_size;
  const float* x     = (const float*)d_in[0];
  const float* mask  = (const float*)d_in[1];
  const float* w_qkv = (const float*)d_in[2];
  const float* b_qkv = (const float*)d_in[3];
  const float* w_out = (const float*)d_in[4];
  const float* b_out = (const float*)d_in[5];
  const float* ebias = (const float*)d_in[6];

  dim3 grid(NLAT * TOW);     // 3720 windows, one block each
  dim3 block(NT);            // 9 waves: one 16-row strip per wave
  hipLaunchKernelGGL(earth_attn3d_kernel, grid, block, 0, stream,
                     x, mask, w_qkv, b_qkv, w_out, b_out, ebias, (float*)d_out);
}
